// MultiHeadSelfAttention_29420525978162
// MI455X (gfx1250) — matrix operand order repaired and checked by reading, not compiled, not run on hardware
//
#include <hip/hip_runtime.h>
#include <math.h>

// ---------------------------------------------------------------------------
// MI455X (gfx1250) multi-head self-attention with RoPE.
// All GEMMs + attention run on v_wmma_f32_16x16x32_f16 (wave32), f32 accum.
// Round-5 change: __builtin_amdgcn_sched_barrier(0) between load groups and
// WMMA groups. Round 4 showed the scheduler sinking each B load to its single
// use, letting RA recycle one 8-VGPR buffer for all four B fragments ->
// s_wait_loadcnt 0x0 before every WMMA. The barrier keeps all fragment loads
// of a (half-)step in flight together, forcing distinct buffers + staggered
// partial waits so the matrix pipe overlaps the load stream.
// ---------------------------------------------------------------------------

typedef _Float16 f16;
typedef _Float16 v16h __attribute__((ext_vector_type(16)));
typedef float    v8f  __attribute__((ext_vector_type(8)));

#define D_MODEL   1024
#define NUM_HEADS 16
#define DK        64
#define SEQ       2048
#define BATCH     2
#define MROWS     (BATCH * SEQ)   // 4096 flattened (b, s) rows

// Two 16-byte chunks -> one A-style v16h fragment.
__device__ __forceinline__ v16h ld16h_2x8(const f16* p0, const f16* p1) {
  union { v16h v; uint4 u[2]; } r;
  r.u[0] = *(const uint4*)p0;
  r.u[1] = *(const uint4*)p1;
  return r.v;
}
// One contiguous 32-byte chunk -> B-style v16h fragment.
__device__ __forceinline__ v16h ld16h(const f16* p) {
  union { v16h v; uint4 u[2]; } r;
  r.u[0] = ((const uint4*)p)[0];
  r.u[1] = ((const uint4*)p)[1];
  return r.v;
}

__device__ __forceinline__ v8f wmma_f16(v16h a, v16h b, v8f c) {
  return __builtin_amdgcn_wmma_f32_16x16x32_f16(
      /*neg_a=*/false, a, /*neg_b=*/false, b,
      /*c_mod=*/(short)0, c, /*reuse_a=*/false, /*reuse_b=*/false);
}

// ---------------------------------------------------------------------------
// Kernel 1: f32 -> f16 conversion, 4 elements/thread (b128 in, b64 out).
// ---------------------------------------------------------------------------
__global__ void cvt_f32_f16(const float* __restrict__ src, f16* __restrict__ dst, int n) {
  int i = (blockIdx.x * blockDim.x + threadIdx.x) * 4;
  if (i < n) {
    float4 v = *(const float4*)(src + i);
    union { f16 h[4]; uint2 u; } r;
    r.h[0] = (f16)v.x; r.h[1] = (f16)v.y; r.h[2] = (f16)v.z; r.h[3] = (f16)v.w;
    *(uint2*)(dst + i) = r.u;
  }
}

// ---------------------------------------------------------------------------
// 16x64 GEMM core, 2x-unrolled ping-pong on A, sched_barrier-pipelined:
// each half-step issues 8 B loads + 2 A loads (next half-step), then a
// scheduling fence, then 4 independent WMMAs -> staggered loadcnt waits.
// ---------------------------------------------------------------------------
__device__ __forceinline__ void gemm16x64(const f16* __restrict__ arow,
                                          const f16* __restrict__ wrow,
                                          v8f acc[4]) {
  v16h a0 = ld16h_2x8(arow, arow + 8);
  v16h a1;
#pragma unroll 1
  for (int k0 = 0; k0 < D_MODEL; k0 += 64) {
    {
      v16h b0 = ld16h(wrow + k0);
      v16h b1 = ld16h(wrow + k0 + 16 * D_MODEL);
      v16h b2 = ld16h(wrow + k0 + 32 * D_MODEL);
      v16h b3 = ld16h(wrow + k0 + 48 * D_MODEL);
      a1 = ld16h_2x8(arow + k0 + 32, arow + k0 + 40);
      __builtin_amdgcn_sched_barrier(0);
      acc[0] = wmma_f16(a0, b0, acc[0]);
      acc[1] = wmma_f16(a0, b1, acc[1]);
      acc[2] = wmma_f16(a0, b2, acc[2]);
      acc[3] = wmma_f16(a0, b3, acc[3]);
    }
    {
      const int kn = (k0 + 64 < D_MODEL) ? k0 + 64 : k0 + 32;  // clamp last
      v16h b0 = ld16h(wrow + k0 + 32);
      v16h b1 = ld16h(wrow + k0 + 32 + 16 * D_MODEL);
      v16h b2 = ld16h(wrow + k0 + 32 + 32 * D_MODEL);
      v16h b3 = ld16h(wrow + k0 + 32 + 48 * D_MODEL);
      a0 = ld16h_2x8(arow + kn, arow + kn + 8);
      __builtin_amdgcn_sched_barrier(0);
      acc[0] = wmma_f16(a1, b0, acc[0]);
      acc[1] = wmma_f16(a1, b1, acc[1]);
      acc[2] = wmma_f16(a1, b2, acc[2]);
      acc[3] = wmma_f16(a1, b3, acc[3]);
    }
  }
}

// ---------------------------------------------------------------------------
// Kernel 2: QKV projection (x @ Wqkv^T) + RoPE epilogue.
// One wave -> 16x64 tile; the 64 columns are exactly one head of {Q,K,V}.
// Q,K stored row-major [b,h,s,dk] after RoPE; V stored transposed [b,h,dk,s].
// ---------------------------------------------------------------------------
__global__ void __launch_bounds__(32)
qkv_gemm_rope(const f16* __restrict__ xh, const f16* __restrict__ wqkv,
              const int* __restrict__ tpos,
              f16* __restrict__ qh, f16* __restrict__ kh,
              f16* __restrict__ vt) {
  const int lane = threadIdx.x;
  const int lo = lane & 15, hi = lane >> 4;
  const int rbase = blockIdx.x * 16;          // flattened (b,s) row tile
  const int jbase = blockIdx.y * 64;          // output feature group
  const int mat = jbase >> 10;                // 0=Q 1=K 2=V
  const int h   = (jbase & 1023) >> 6;        // head
  const int b   = rbase / SEQ;

  v8f acc[4] = {};
  gemm16x64(xh   + (size_t)(rbase + lo) * D_MODEL + hi * 16,
            wqkv + (size_t)(jbase + lo) * D_MODEL + hi * 16, acc);

#pragma unroll
  for (int nt = 0; nt < 4; ++nt) {
#pragma unroll
    for (int r = 0; r < 8; ++r) {
      const int srow = rbase + r + 8 * hi;     // C layout: reg r -> row r / r+8
      const int s  = srow - b * SEQ;
      const int dd = nt * 16 + lo;             // col within head (0..63)
      float v = acc[nt][r];
      if (mat < 2) {
        // RoPE: pairs (even,odd) are adjacent dd -> adjacent lanes.
        const float pos  = (float)tpos[s];
        const float invf = __expf(-(float)(2 * (dd >> 1)) * (9.210340371976184f / 64.0f));
        float sn, cs;
        __sincosf(pos * invf, &sn, &cs);
        const float partner = __shfl_xor(v, 1, 32);
        const float y = (dd & 1) ? (v * cs + partner * sn) : (v * cs - partner * sn);
        f16* dst = (mat == 0) ? qh : kh;
        dst[((size_t)(b * NUM_HEADS + h) * SEQ + s) * DK + dd] = (f16)y;
      } else {
        vt[((size_t)(b * NUM_HEADS + h) * DK + dd) * SEQ + s] = (f16)v;
      }
    }
  }
}

// ---------------------------------------------------------------------------
// Kernel 3: causal flash attention. One wave per (b*h, 16-query tile).
// Keys in chunks of 32. Fragment loads grouped ahead of WMMA groups with
// sched_barrier; V/next-K lines cache-warmed with global_prefetch.
// ---------------------------------------------------------------------------
__global__ void __launch_bounds__(32)
attn_flash(const f16* __restrict__ qh, const f16* __restrict__ kh,
           const f16* __restrict__ vt, f16* __restrict__ attnh) {
  __shared__ f16 ldsP[16 * 32];                // P tile, C-layout -> A-layout bounce
  const int lane = threadIdx.x;
  const int lo = lane & 15, hi = lane >> 4;
  const int bh = blockIdx.y;                   // 0..31
  const int q0 = blockIdx.x * 16;
  const int b  = bh >> 4;

  // Q tile as two A fragments (dk 0..31, 32..63)
  const f16* qb = qh + ((size_t)bh * SEQ + q0 + lo) * DK;
  const v16h aq0 = ld16h_2x8(qb + hi * 16,      qb + hi * 16 + 8);
  const v16h aq1 = ld16h_2x8(qb + 32 + hi * 16, qb + 32 + hi * 16 + 8);

  float m[8], l[8];
#pragma unroll
  for (int r = 0; r < 8; ++r) { m[r] = -__builtin_inff(); l[r] = 0.0f; }
  v8f acc[4] = {};

  const int lastC = q0 >> 5;                   // diagonal chunk; skip above it
#pragma unroll 1
  for (int ch = 0; ch <= lastC; ++ch) {
    const int c = ch * 32;

    // --- scores = Q @ K^T: load all 4 K fragments, fence, then 4 WMMAs ---
    const f16* kb0 = kh + ((size_t)bh * SEQ + c + lo) * DK + hi * 16;
    const f16* kb1 = kh + ((size_t)bh * SEQ + c + 16 + lo) * DK + hi * 16;
    v16h k00 = ld16h(kb0);
    v16h k01 = ld16h(kb0 + 32);
    v16h k10 = ld16h(kb1);
    v16h k11 = ld16h(kb1 + 32);
    __builtin_amdgcn_sched_barrier(0);
    v8f s[2];
    {
      v8f z0 = {}, z1 = {};
      z0 = wmma_f16(aq0, k00, z0);             // interleave the two independent
      z1 = wmma_f16(aq0, k10, z1);             // accumulator chains
      z0 = wmma_f16(aq1, k01, z0);
      z1 = wmma_f16(aq1, k11, z1);
      s[0] = z0; s[1] = z1;
    }

    // --- cache-warm V for this chunk and K for the next (global_prefetch) ---
#pragma unroll
    for (int n = 0; n < 4; ++n)
      __builtin_prefetch(vt + ((size_t)bh * DK + n * 16 + lo) * SEQ + c + hi * 16, 0, 3);
    if (ch < lastC) {
      __builtin_prefetch(kh + ((size_t)bh * SEQ + c + 32 + lo) * DK + hi * 16, 0, 3);
      __builtin_prefetch(kh + ((size_t)bh * SEQ + c + 48 + lo) * DK + hi * 16, 0, 3);
    }

    // --- online softmax ---
    const bool diag = (ch == lastC);
#pragma unroll
    for (int t = 0; t < 2; ++t)
#pragma unroll
      for (int r = 0; r < 8; ++r) {
        float v = s[t][r] * 0.125f;            // 1/sqrt(64)
        if (diag) {
          const int key = c + 16 * t + lo;
          const int qq  = q0 + r + 8 * hi;
          if (key > qq) v = -__builtin_inff();
        }
        s[t][r] = v;
      }

#pragma unroll
    for (int r = 0; r < 8; ++r) {
      float rm = fmaxf(s[0][r], s[1][r]);      // row max within 16-lane halves
      rm = fmaxf(rm, __shfl_xor(rm, 1, 32));
      rm = fmaxf(rm, __shfl_xor(rm, 2, 32));
      rm = fmaxf(rm, __shfl_xor(rm, 4, 32));
      rm = fmaxf(rm, __shfl_xor(rm, 8, 32));
      const float mnew = fmaxf(m[r], rm);
      const float sc = __expf(m[r] - mnew);
      m[r] = mnew;
      float rs = 0.0f;
#pragma unroll
      for (int t = 0; t < 2; ++t) {
        const float p = __expf(s[t][r] - mnew);
        s[t][r] = p;
        rs += p;
      }
      rs += __shfl_xor(rs, 1, 32);
      rs += __shfl_xor(rs, 2, 32);
      rs += __shfl_xor(rs, 4, 32);
      rs += __shfl_xor(rs, 8, 32);
      l[r] = l[r] * sc + rs;
      // rescale the output accumulators for this row
#pragma unroll
      for (int n = 0; n < 4; ++n) acc[n][r] *= sc;
    }

    // --- P: C-layout -> LDS (16x32 f16) -> A-layout fragment ---
#pragma unroll
    for (int t = 0; t < 2; ++t)
#pragma unroll
      for (int r = 0; r < 8; ++r)
        ldsP[(r + 8 * hi) * 32 + 16 * t + lo] = (f16)s[t][r];
    asm volatile("s_wait_dscnt 0x0" ::: "memory");

    // --- PV: load P fragment + all 4 V fragments, fence, then 4 WMMAs ---
    const v16h ap = ld16h_2x8(&ldsP[lo * 32 + hi * 16], &ldsP[lo * 32 + hi * 16 + 8]);
    v16h bv0 = ld16h(vt + ((size_t)bh * DK +  0 + lo) * SEQ + c + hi * 16);
    v16h bv1 = ld16h(vt + ((size_t)bh * DK + 16 + lo) * SEQ + c + hi * 16);
    v16h bv2 = ld16h(vt + ((size_t)bh * DK + 32 + lo) * SEQ + c + hi * 16);
    v16h bv3 = ld16h(vt + ((size_t)bh * DK + 48 + lo) * SEQ + c + hi * 16);
    __builtin_amdgcn_sched_barrier(0);
    acc[0] = wmma_f16(ap, bv0, acc[0]);
    acc[1] = wmma_f16(ap, bv1, acc[1]);
    acc[2] = wmma_f16(ap, bv2, acc[2]);
    acc[3] = wmma_f16(ap, bv3, acc[3]);
  }

  const int h = bh & 15;
#pragma unroll
  for (int r = 0; r < 8; ++r) {
    const float inv = 1.0f / l[r];
    const int srow = b * SEQ + q0 + r + 8 * hi;
#pragma unroll
    for (int n = 0; n < 4; ++n)
      attnh[(size_t)srow * D_MODEL + h * DK + n * 16 + lo] = (f16)(acc[n][r] * inv);
  }
}

// ---------------------------------------------------------------------------
// Kernel 4: output projection  out = attn @ WO^T  (f32 output)
// ---------------------------------------------------------------------------
__global__ void __launch_bounds__(32)
out_gemm(const f16* __restrict__ ah, const f16* __restrict__ woh,
         float* __restrict__ out) {
  const int lane = threadIdx.x;
  const int lo = lane & 15, hi = lane >> 4;
  const int rbase = blockIdx.x * 16;
  const int jbase = blockIdx.y * 64;

  v8f acc[4] = {};
  gemm16x64(ah  + (size_t)(rbase + lo) * D_MODEL + hi * 16,
            woh + (size_t)(jbase + lo) * D_MODEL + hi * 16, acc);

#pragma unroll
  for (int nt = 0; nt < 4; ++nt)
#pragma unroll
    for (int r = 0; r < 8; ++r)
      out[(size_t)(rbase + r + 8 * hi) * D_MODEL + jbase + nt * 16 + lo] = acc[nt][r];
}

// ---------------------------------------------------------------------------
extern "C" void kernel_launch(void* const* d_in, const int* in_sizes, int n_in,
                              void* d_out, int out_size, void* d_ws, size_t ws_size,
                              hipStream_t stream) {
  (void)in_sizes; (void)n_in; (void)out_size; (void)ws_size;
  const float* x    = (const float*)d_in[0];
  const int*   tpos = (const int*)d_in[1];
  const float* WQ   = (const float*)d_in[2];
  const float* WK   = (const float*)d_in[3];
  const float* WV   = (const float*)d_in[4];
  const float* WO   = (const float*)d_in[5];
  float* out = (float*)d_out;

  // Workspace layout (40 MB total; attnh aliases xh which is dead by then):
  char* ws = (char*)d_ws;
  const size_t MB = 1024u * 1024u;
  f16* xh    = (f16*)(ws + 0);          // 8 MB  [4096,1024] f16
  f16* wqkvh = (f16*)(ws + 8 * MB);     // 6 MB  [3072,1024] f16 (WQ;WK;WV rows)
  f16* woh   = (f16*)(ws + 14 * MB);    // 2 MB  [1024,1024] f16
  f16* qh    = (f16*)(ws + 16 * MB);    // 8 MB  [b,h,s,dk]
  f16* kh    = (f16*)(ws + 24 * MB);    // 8 MB  [b,h,s,dk]
  f16* vt    = (f16*)(ws + 32 * MB);    // 8 MB  [b,h,dk,s] (transposed)
  f16* attnh = xh;                      // alias: [4096,1024] f16

  const int nX = MROWS * D_MODEL;       // 4M
  const int nW = D_MODEL * D_MODEL;     // 1M
  cvt_f32_f16<<<(nX / 4 + 255) / 256, 256, 0, stream>>>(x, xh, nX);
  cvt_f32_f16<<<(nW / 4 + 255) / 256, 256, 0, stream>>>(WQ, wqkvh, nW);
  cvt_f32_f16<<<(nW / 4 + 255) / 256, 256, 0, stream>>>(WK, wqkvh + (size_t)nW, nW);
  cvt_f32_f16<<<(nW / 4 + 255) / 256, 256, 0, stream>>>(WV, wqkvh + (size_t)2 * nW, nW);
  cvt_f32_f16<<<(nW / 4 + 255) / 256, 256, 0, stream>>>(WO, woh, nW);

  qkv_gemm_rope<<<dim3(MROWS / 16, 48), 32, 0, stream>>>(xh, wqkvh, tpos, qh, kh, vt);
  attn_flash<<<dim3(SEQ / 16, BATCH * NUM_HEADS), 32, 0, stream>>>(qh, kh, vt, attnh);
  out_gemm<<<dim3(MROWS / 16, 16), 32, 0, stream>>>(attnh, woh, out);
}
